// MoEBlock_40407052320888
// MI455X (gfx1250) — matrix operand order repaired and checked by reading, not compiled, not run on hardware
//
#include <hip/hip_runtime.h>
#include <hip/hip_bf16.h>

// ---------------- problem constants (from reference setup_inputs) -------------
#define NTOK   8192     // B*S = 4*2048
#define DDIM   2048     // model dim D
#define EXP    32       // experts E
#define IDIM   1024     // expert intermediate I
#define TOPK   8
#define ROUTED_SCALE 2.5f

typedef __attribute__((ext_vector_type(16))) __bf16 v16bf;
typedef __attribute__((ext_vector_type(8)))  __bf16 v8bf;
typedef __attribute__((ext_vector_type(4)))  __bf16 v4bf;
typedef __attribute__((ext_vector_type(8)))  float  v8f;
typedef __attribute__((ext_vector_type(4)))  unsigned int u32x4;
typedef __attribute__((ext_vector_type(4)))  int i32x4;
typedef __attribute__((ext_vector_type(8)))  int i32x8;

#if __has_builtin(__builtin_amdgcn_tensor_load_to_lds) && \
    __has_builtin(__builtin_amdgcn_s_wait_tensorcnt)
#define HAVE_TDM 1
#endif

static __device__ __forceinline__ __bf16 f2bf(float f) {
  unsigned u = __builtin_bit_cast(unsigned, f);
  unsigned r = u + 0x7FFFu + ((u >> 16) & 1u);          // round-to-nearest-even
  unsigned short h = (unsigned short)(r >> 16);
  return __builtin_bit_cast(__bf16, h);
}

static __device__ __forceinline__ float fast_sigmoid(float g) {
  return __builtin_amdgcn_rcpf(1.f + __expf(-g));       // v_rcp_f32 + v_exp_f32
}

static __device__ __forceinline__ v16bf bfcat(v8bf lo, v8bf hi) {
  return __builtin_shufflevector(lo, hi, 0,1,2,3,4,5,6,7,8,9,10,11,12,13,14,15);
}

static __device__ __forceinline__ v8f wmma_bf16(v16bf a, v16bf b, v8f c) {
  return __builtin_amdgcn_wmma_f32_16x16x32_bf16(false, a, false, b, (short)0, c,
                                                 false, false);
}

#ifdef HAVE_TDM
// Issue one TDM 2-D tile load: tile_n rows x tile_k bf16 elements from a
// row-major [rows][rowlen] bf16 tensor into LDS (rows packed back-to-back).
// D# packing per CDNA5 ISA 8.3/8.4: count=1, type=2, data_size=1 (2 bytes).
static __device__ __forceinline__ void tdm_load_2d(unsigned lds_off,
                                                   const void* gaddr,
                                                   unsigned rowlen,      // tensor_dim0 (elts)
                                                   unsigned nrows,       // tensor_dim1
                                                   unsigned tile_k,      // tile_dim0 (elts)
                                                   unsigned tile_n) {    // tile_dim1
  unsigned long long ga = (unsigned long long)(uintptr_t)gaddr;
  u32x4 g0;
  g0[0] = 1u;                                            // count=1, user mode
  g0[1] = lds_off;                                       // lds_addr (bytes)
  g0[2] = (unsigned)ga;                                  // global_addr[31:0]
  g0[3] = (unsigned)((ga >> 32) & 0x1FFFFFFu) | (2u << 30); // addr[56:32] | type=2
  i32x8 g1;
  g1[0] = (int)(1u << 16);                               // data_size=1 (2B)
  g1[1] = (int)((rowlen & 0xFFFFu) << 16);               // tensor_dim0[15:0]
  g1[2] = (int)((rowlen >> 16) | ((nrows & 0xFFFFu) << 16));
  g1[3] = (int)((nrows >> 16) | (tile_k << 16));         // tile_dim0
  g1[4] = (int)tile_n;                                   // tile_dim1, tile_dim2=0
  g1[5] = (int)rowlen;                                   // tensor_dim0_stride[31:0]
  g1[6] = 0;
  g1[7] = 0;
  i32x4 z4 = {0, 0, 0, 0};
#if __clang_major__ >= 23
  i32x8 z8 = {0, 0, 0, 0, 0, 0, 0, 0};
  __builtin_amdgcn_tensor_load_to_lds(g0, g1, z4, z4, z8, 0);
#else
  __builtin_amdgcn_tensor_load_to_lds(g0, g1, z4, z4, 0);
#endif
}
#endif

// ---------------------------------------------------------------------------
// x (f32) -> bf16
__global__ __launch_bounds__(256) void cvt_kernel(const float* __restrict__ x,
                                                  __bf16* __restrict__ xb, int n4) {
  int i = blockIdx.x * blockDim.x + threadIdx.x;
  if (i < n4) {
    float4 v = ((const float4*)x)[i];
    v4bf o;
    o[0] = f2bf(v.x); o[1] = f2bf(v.y); o[2] = f2bf(v.z); o[3] = f2bf(v.w);
    ((v4bf*)xb)[i] = o;
  }
}

// ---------------------------------------------------------------------------
// Transpose + convert: f32 [K][N] row-major  ->  bf16 [N][K] row-major.
// blockIdx.z selects the matrix (expert).
__global__ __launch_bounds__(256)
void tcvt_kernel(const float* __restrict__ src, __bf16* __restrict__ dst,
                 int K, int N) {
  __shared__ float ts[64][65];
  const float* s = src + (size_t)blockIdx.z * K * N;
  __bf16*      d = dst + (size_t)blockIdx.z * K * N;
  const int n0 = blockIdx.x * 64, k0 = blockIdx.y * 64;
  const int t = threadIdx.x;
  {
    int kk = t >> 2, nn = (t & 3) << 4;
#pragma unroll
    for (int i = 0; i < 16; i += 4) {
      float4 v = *(const float4*)(s + (size_t)(k0 + kk) * N + (n0 + nn + i));
      ts[kk][nn + i + 0] = v.x; ts[kk][nn + i + 1] = v.y;
      ts[kk][nn + i + 2] = v.z; ts[kk][nn + i + 3] = v.w;
    }
  }
  __syncthreads();
  {
    int on = t >> 2, ok = (t & 3) << 4;
    alignas(16) __bf16 tmp[16];
#pragma unroll
    for (int i = 0; i < 16; ++i) tmp[i] = f2bf(ts[ok + i][on]);
    __bf16* p = d + (size_t)(n0 + on) * K + k0 + ok;
    *(int4*)p       = *(int4*)&tmp[0];
    *(int4*)(p + 8) = *(int4*)&tmp[8];
  }
}

// ---------------------------------------------------------------------------
// Router: one wave per token; lane == expert (E == 32 == wave32 width).
__global__ __launch_bounds__(256)
void router_kernel(const float* __restrict__ x, const float* __restrict__ gw,
                   const float* __restrict__ ebias,
                   float* __restrict__ scores_out, int* __restrict__ idx_out,
                   float* __restrict__ topkw_out, int* __restrict__ cnt) {
  __shared__ float sm_sfr[8][32];
  __shared__ float sm_s[8][32];
  const int wave = threadIdx.x >> 5, lane = threadIdx.x & 31;
  const int tok  = blockIdx.x * 8 + wave;
  const float* xr = x + (size_t)tok * DDIM;
  float acc = 0.f;
  for (int d = 0; d < DDIM; d += 4) {
    float4 xv = *(const float4*)(xr + d);
    acc += xv.x * gw[(d + 0) * EXP + lane];
    acc += xv.y * gw[(d + 1) * EXP + lane];
    acc += xv.z * gw[(d + 2) * EXP + lane];
    acc += xv.w * gw[(d + 3) * EXP + lane];
  }
  float s = fast_sigmoid(acc);
  scores_out[(size_t)tok * EXP + lane] = s;
  sm_s[wave][lane]   = s;
  sm_sfr[wave][lane] = s + ebias[lane];
  __syncthreads();
  if (lane == 0) {
    float* sfr = sm_sfr[wave];
    float* sc  = sm_s[wave];
    float gs[8];
    for (int g = 0; g < 8; ++g) {              // sum of top-2 of each group of 4
      float m1 = -1e30f, m2 = -1e30f;
      for (int j = 0; j < 4; ++j) {
        float v = sfr[g * 4 + j];
        if (v > m1) { m2 = m1; m1 = v; } else if (v > m2) m2 = v;
      }
      gs[g] = m1 + m2;
    }
    bool gsel[8] = {false,false,false,false,false,false,false,false};
    for (int it = 0; it < 4; ++it) {           // top-4 groups
      int best = 0; float bv = -1e30f;
      for (int g = 0; g < 8; ++g) if (!gsel[g] && gs[g] > bv) { bv = gs[g]; best = g; }
      gsel[best] = true;
    }
    float masked[32];
    for (int e = 0; e < EXP; ++e) masked[e] = gsel[e >> 2] ? sfr[e] : -1e30f;
    int kid[TOPK]; float kw[TOPK]; float wsum = 0.f;
    for (int kk = 0; kk < TOPK; ++kk) {        // top-8 experts
      int best = 0; float bv = -2e30f;
      for (int e = 0; e < EXP; ++e) if (masked[e] > bv) { bv = masked[e]; best = e; }
      masked[best] = -3e30f;
      kid[kk] = best; kw[kk] = sc[best]; wsum += sc[best];
    }
    float inv = ROUTED_SCALE * __builtin_amdgcn_rcpf(wsum + 1e-20f);
    for (int kk = 0; kk < TOPK; ++kk) {
      idx_out[(size_t)tok * TOPK + kk]   = kid[kk];
      topkw_out[(size_t)tok * TOPK + kk] = kw[kk] * inv;
      atomicAdd(&cnt[kid[kk]], 1);
    }
  }
}

__global__ void scan_kernel(int* __restrict__ cnt, int* __restrict__ off,
                            int* __restrict__ cur) {
  if (threadIdx.x == 0) {
    int acc = 0;
    for (int e = 0; e < EXP; ++e) { off[e] = acc; cur[e] = acc; acc += cnt[e]; }
  }
}

__global__ __launch_bounds__(256)
void scatter_kernel(const int* __restrict__ idx, const float* __restrict__ tw,
                    int* __restrict__ cur, int* __restrict__ rtok,
                    float* __restrict__ rw) {
  int i = blockIdx.x * blockDim.x + threadIdx.x;   // < NTOK*TOPK
  int e = idx[i];
  int pos = atomicAdd(&cur[e], 1);
  rtok[pos] = i >> 3;
  rw[pos]   = tw[i];
}

// ---------------------------------------------------------------------------
// Pass 1: inter = silu(x@Wg) * (x@Wu), bf16 out. 64x64 tile, K-step 32.
// Double-buffered LDS: TDM(k+1) + A-gather(k+1) overlap compute(k).
template <bool ROUTED>
__global__ __launch_bounds__(256)
void pass1_kernel(const __bf16* __restrict__ xb,
                  const __bf16* __restrict__ BgT_base,   // bf16 [n][DDIM]
                  const __bf16* __restrict__ BuT_base,
                  __bf16* __restrict__ inter,
                  const int* __restrict__ row_tok,
                  const int* __restrict__ cnt, const int* __restrict__ off) {
  __shared__ __bf16 Al[2][64 * 32];    // [m][k]
  __shared__ __bf16 Bgl[2][64 * 32];   // [n][k]
  __shared__ __bf16 Bul[2][64 * 32];

  const int tid = threadIdx.x;
  const int wave = tid >> 5, lane = tid & 31, half = lane >> 4, lm = lane & 15;
  const int wm = wave & 3, wn = wave >> 2;

  const __bf16 *BgT, *BuT;
  int rstart, rcount, n0;
  if (ROUTED) {
    int e = blockIdx.x >> 4;
    n0 = (blockIdx.x & 15) * 64;
    rstart = off[e]; rcount = cnt[e];
    BgT = BgT_base + (size_t)e * (2 * IDIM) * DDIM;
    BuT = BgT + (size_t)IDIM * DDIM;
  } else {
    n0 = (blockIdx.x & 15) * 64;
    rstart = (blockIdx.x >> 4) * 64; rcount = 64;
    BgT = BgT_base; BuT = BuT_base;
  }

  const v8f vzero = {0.f,0.f,0.f,0.f,0.f,0.f,0.f,0.f};
  const int sr = tid >> 2, scc = (tid & 3) << 3;   // staging row / col

  for (int m0 = 0; m0 < rcount; m0 += 64) {
    // hoist the A-row gather for this m-tile out of the K loop
    const bool avalid = (m0 + sr) < rcount;
    const int  arow   = ROUTED ? (avalid ? row_tok[rstart + m0 + sr] : 0)
                               : (rstart + m0 + sr);
    const __bf16* abase = xb + (size_t)arow * DDIM + scc;

    auto stage = [&](int k0, int buf) {
#ifdef HAVE_TDM
      if (wave == 0) {
        tdm_load_2d((unsigned)(uintptr_t)(void*)&Bgl[buf][0],
                    BgT + (size_t)n0 * DDIM + k0, DDIM, IDIM, 32, 64);
        tdm_load_2d((unsigned)(uintptr_t)(void*)&Bul[buf][0],
                    BuT + (size_t)n0 * DDIM + k0, DDIM, IDIM, 32, 64);
      }
#else
      *(int4*)(&Bgl[buf][sr * 32 + scc]) =
          *(const int4*)(BgT + (size_t)(n0 + sr) * DDIM + k0 + scc);
      *(int4*)(&Bul[buf][sr * 32 + scc]) =
          *(const int4*)(BuT + (size_t)(n0 + sr) * DDIM + k0 + scc);
#endif
      int4 av;
      if (avalid) av = *(const int4*)(abase + k0);
      else { av.x = 0; av.y = 0; av.z = 0; av.w = 0; }
      if (k0 + 64 < DDIM) __builtin_prefetch(abase + k0 + 64, 0, 3);
      *(int4*)(&Al[buf][sr * 32 + scc]) = av;
    };
    auto wait_tiles = [&]() {
#ifdef HAVE_TDM
      if (wave == 0) __builtin_amdgcn_s_wait_tensorcnt(0);
#endif
    };

    v8f accg[2] = {vzero, vzero};
    v8f accu[2] = {vzero, vzero};

    stage(0, 0);
    wait_tiles();
    __syncthreads();

    for (int k0 = 0; k0 < DDIM; k0 += 32) {
      const int cur = (k0 >> 5) & 1, nxt = cur ^ 1;
      const bool more = (k0 + 32) < DDIM;
      if (more) stage(k0 + 32, nxt);
      // ---- compute on buffer `cur`: load all fragments first, then WMMA ----
      {
        const int arow_l = (wm * 16 + lm) * 32;
        v8bf a0 = *(const v8bf*)&Al[cur][arow_l + half * 8];
        v8bf a1 = *(const v8bf*)&Al[cur][arow_l + 16 + half * 8];
        v8bf bg0[2], bg1[2], bu0[2], bu1[2];
#pragma unroll
        for (int nt = 0; nt < 2; ++nt) {
          const int nrow = (wn * 32 + nt * 16 + lm) * 32 + half * 8;
          bg0[nt] = *(const v8bf*)&Bgl[cur][nrow];
          bg1[nt] = *(const v8bf*)&Bgl[cur][nrow + 16];
          bu0[nt] = *(const v8bf*)&Bul[cur][nrow];
          bu1[nt] = *(const v8bf*)&Bul[cur][nrow + 16];
        }
        v16bf av = bfcat(a0, a1);
#pragma unroll
        for (int nt = 0; nt < 2; ++nt) {
          accg[nt] = wmma_bf16(av, bfcat(bg0[nt], bg1[nt]), accg[nt]);
          accu[nt] = wmma_bf16(av, bfcat(bu0[nt], bu1[nt]), accu[nt]);
        }
      }
      if (more) wait_tiles();
      __syncthreads();
    }
    // ---- epilogue: SwiGLU, store bf16 ----
#pragma unroll
    for (int nt = 0; nt < 2; ++nt) {
      int col = n0 + wn * 32 + nt * 16 + lm;
#pragma unroll
      for (int rr = 0; rr < 8; ++rr) {
        int m = wm * 16 + (half << 3) + rr;
        if (m0 + m < rcount) {
          float g = accg[nt][rr], u = accu[nt][rr];
          float val = g * fast_sigmoid(g) * u;
          inter[(size_t)(rstart + m0 + m) * IDIM + col] = f2bf(val);
        }
      }
    }
  }
}

// ---------------------------------------------------------------------------
// Pass 2: out = inter @ Wd ; routed: weighted atomic add into d_out.
// Both A and B tiles via TDM, double-buffered LDS.
template <bool ROUTED>
__global__ __launch_bounds__(256)
void pass2_kernel(const __bf16* __restrict__ Ain, const __bf16* __restrict__ WdT_base,
                  float* __restrict__ outp,
                  const int* __restrict__ row_tok, const float* __restrict__ row_w,
                  const int* __restrict__ cnt, const int* __restrict__ off) {
  __shared__ __bf16 Al[2][64 * 32];
  __shared__ __bf16 Bl[2][64 * 32];

  const int tid = threadIdx.x;
  const int wave = tid >> 5, lane = tid & 31, half = lane >> 4, lm = lane & 15;
  const int wm = wave & 3, wn = wave >> 2;

  const __bf16* WdT;
  int rstart, rcount, n0;
  if (ROUTED) {
    int e = blockIdx.x >> 5;
    n0 = (blockIdx.x & 31) * 64;
    rstart = off[e]; rcount = cnt[e];
    WdT = WdT_base + (size_t)e * DDIM * IDIM;
  } else {
    n0 = (blockIdx.x & 31) * 64;
    rstart = (blockIdx.x >> 5) * 64; rcount = 64;
    WdT = WdT_base;
  }

  const v8f vzero = {0.f,0.f,0.f,0.f,0.f,0.f,0.f,0.f};
  const int sr = tid >> 2, scc = (tid & 3) << 3;

  for (int m0 = 0; m0 < rcount; m0 += 64) {
    const __bf16* abase = Ain + (size_t)(rstart + m0) * IDIM;

    auto stage = [&](int k0, int buf) {
#ifdef HAVE_TDM
      if (wave == 0) {
        tdm_load_2d((unsigned)(uintptr_t)(void*)&Al[buf][0],
                    abase + k0, IDIM, 1u << 20, 32, 64);
        tdm_load_2d((unsigned)(uintptr_t)(void*)&Bl[buf][0],
                    WdT + (size_t)n0 * IDIM + k0, IDIM, DDIM, 32, 64);
      }
#else
      {
        bool valid = (m0 + sr) < rcount;
        int4 av;
        if (valid) av = *(const int4*)(abase + (size_t)sr * IDIM + k0 + scc);
        else { av.x = 0; av.y = 0; av.z = 0; av.w = 0; }
        *(int4*)(&Al[buf][sr * 32 + scc]) = av;
        *(int4*)(&Bl[buf][sr * 32 + scc]) =
            *(const int4*)(WdT + (size_t)(n0 + sr) * IDIM + k0 + scc);
      }
#endif
    };
    auto wait_tiles = [&]() {
#ifdef HAVE_TDM
      if (wave == 0) __builtin_amdgcn_s_wait_tensorcnt(0);
#endif
    };

    v8f acc[2] = {vzero, vzero};

    stage(0, 0);
    wait_tiles();
    __syncthreads();

    for (int k0 = 0; k0 < IDIM; k0 += 32) {
      const int cur = (k0 >> 5) & 1, nxt = cur ^ 1;
      const bool more = (k0 + 32) < IDIM;
      if (more) stage(k0 + 32, nxt);
      {
        const int arow_l = (wm * 16 + lm) * 32;
        v8bf a0 = *(const v8bf*)&Al[cur][arow_l + half * 8];
        v8bf a1 = *(const v8bf*)&Al[cur][arow_l + 16 + half * 8];
        v8bf b0[2], b1[2];
#pragma unroll
        for (int nt = 0; nt < 2; ++nt) {
          const int nrow = (wn * 32 + nt * 16 + lm) * 32 + half * 8;
          b0[nt] = *(const v8bf*)&Bl[cur][nrow];
          b1[nt] = *(const v8bf*)&Bl[cur][nrow + 16];
        }
        v16bf av = bfcat(a0, a1);
#pragma unroll
        for (int nt = 0; nt < 2; ++nt) {
          acc[nt] = wmma_bf16(av, bfcat(b0[nt], b1[nt]), acc[nt]);
        }
      }
      if (more) wait_tiles();
      __syncthreads();
    }
#pragma unroll
    for (int nt = 0; nt < 2; ++nt) {
      int col = n0 + wn * 32 + nt * 16 + lm;
#pragma unroll
      for (int rr = 0; rr < 8; ++rr) {
        int m = wm * 16 + (half << 3) + rr;
        if (m0 + m < rcount) {
          int srow = rstart + m0 + m;
          float v = acc[nt][rr];
          if (ROUTED) {
            atomicAdd(&outp[(size_t)row_tok[srow] * DDIM + col], v * row_w[srow]);
          } else {
            outp[(size_t)srow * DDIM + col] = v;
          }
        }
      }
    }
  }
}

// ---------------------------------------------------------------------------
extern "C" void kernel_launch(void* const* d_in, const int* in_sizes, int n_in,
                              void* d_out, int out_size, void* d_ws, size_t ws_size,
                              hipStream_t stream) {
  (void)in_sizes; (void)n_in; (void)out_size; (void)ws_size;
  const float* x   = (const float*)d_in[0];
  const float* gw  = (const float*)d_in[1];
  const float* eb  = (const float*)d_in[2];
  const float* gup = (const float*)d_in[3];
  const float* dwn = (const float*)d_in[4];
  const float* sg  = (const float*)d_in[5];
  const float* su  = (const float*)d_in[6];
  const float* sd  = (const float*)d_in[7];

  float* outp       = (float*)d_out;                                    // [N, D] f32
  int*   idx_out    = (int*)(outp + (size_t)NTOK * DDIM);               // [N, 8] i32
  float* scores_out = outp + (size_t)NTOK * DDIM + (size_t)NTOK * TOPK; // [N, 32] f32

  // ---- workspace layout (bytes) ----
  char* wsb = (char*)d_ws;
  int*    cnt   = (int*)wsb;
  int*    off   = cnt + 32;
  int*    cur   = off + 32;
  float*  topkw = (float*)(wsb + 512);                                  // N*K f32
  int*    rtok  = (int*)(wsb + 512 + (size_t)NTOK * TOPK * 4);          // N*K i32
  float*  rw    = (float*)(wsb + 512 + 2 * (size_t)NTOK * TOPK * 4);    // N*K f32
  constexpr size_t OFF_XB  = 1u << 20;
  constexpr size_t SZ_XB   = (size_t)NTOK * DDIM * 2;                   // 32 MB
  constexpr size_t OFF_GUP = OFF_XB + SZ_XB;
  constexpr size_t SZ_GUP  = (size_t)EXP * 2 * IDIM * DDIM * 2;         // 268 MB
  constexpr size_t OFF_DWN = OFF_GUP + SZ_GUP;
  constexpr size_t SZ_DWN  = (size_t)EXP * DDIM * IDIM * 2;             // 134 MB
  constexpr size_t OFF_SG  = OFF_DWN + SZ_DWN;
  constexpr size_t SZ_SGU  = (size_t)IDIM * DDIM * 2;                   // 4 MB each
  constexpr size_t OFF_SU  = OFF_SG + SZ_SGU;
  constexpr size_t OFF_SD  = OFF_SU + SZ_SGU;
  constexpr size_t SZ_SD   = (size_t)DDIM * IDIM * 2;
  constexpr size_t OFF_IR  = OFF_SD + SZ_SD;
  constexpr size_t SZ_IR   = (size_t)NTOK * TOPK * IDIM * 2;            // 128 MB
  constexpr size_t OFF_IS  = OFF_IR + SZ_IR;                            // +16 MB
  __bf16* xb      = (__bf16*)(wsb + OFF_XB);
  __bf16* gupT    = (__bf16*)(wsb + OFF_GUP);  // per expert: [2I n][D k]
  __bf16* dwnT    = (__bf16*)(wsb + OFF_DWN);  // per expert: [D n][I k]
  __bf16* sgT     = (__bf16*)(wsb + OFF_SG);   // [I n][D k]
  __bf16* suT     = (__bf16*)(wsb + OFF_SU);   // [I n][D k]
  __bf16* sdT     = (__bf16*)(wsb + OFF_SD);   // [D n][I k]
  __bf16* inter_r = (__bf16*)(wsb + OFF_IR);
  __bf16* inter_s = (__bf16*)(wsb + OFF_IS);

  hipMemsetAsync(wsb, 0, 512, stream);
  cvt_kernel<<<(NTOK * DDIM / 4 + 255) / 256, 256, 0, stream>>>(x, xb, NTOK * DDIM / 4);
  // one-time weight transpose+convert: f32 [K][N] -> bf16 [N][K]
  tcvt_kernel<<<dim3(2 * IDIM / 64, DDIM / 64, EXP), 256, 0, stream>>>(gup, gupT, DDIM, 2 * IDIM);
  tcvt_kernel<<<dim3(DDIM / 64, IDIM / 64, EXP), 256, 0, stream>>>(dwn, dwnT, IDIM, DDIM);
  tcvt_kernel<<<dim3(IDIM / 64, DDIM / 64, 1), 256, 0, stream>>>(sg, sgT, DDIM, IDIM);
  tcvt_kernel<<<dim3(IDIM / 64, DDIM / 64, 1), 256, 0, stream>>>(su, suT, DDIM, IDIM);
  tcvt_kernel<<<dim3(DDIM / 64, IDIM / 64, 1), 256, 0, stream>>>(sd, sdT, IDIM, DDIM);
  // routing
  router_kernel<<<NTOK / 8, 256, 0, stream>>>(x, gw, eb, scores_out, idx_out, topkw, cnt);
  scan_kernel<<<1, 32, 0, stream>>>(cnt, off, cur);
  scatter_kernel<<<(NTOK * TOPK) / 256, 256, 0, stream>>>(idx_out, topkw, cur, rtok, rw);
  // shared MLP (writes d_out), then routed experts (atomic adds on top)
  pass1_kernel<false><<<(NTOK / 64) * (IDIM / 64), 256, 0, stream>>>(
      xb, sgT, suT, inter_s, nullptr, nullptr, nullptr);
  pass2_kernel<false><<<(NTOK / 64) * (DDIM / 64), 256, 0, stream>>>(
      inter_s, sdT, outp, nullptr, nullptr, nullptr, nullptr);
  pass1_kernel<true><<<EXP * (IDIM / 64), 256, 0, stream>>>(
      xb, gupT, gupT, inter_r, rtok, cnt, off);
  pass2_kernel<true><<<EXP * (DDIM / 64), 256, 0, stream>>>(
      inter_r, dwnT, outp, rtok, rw, cnt, off);
}
